// TokenFormerLayer_38079180047036
// MI455X (gfx1250) — matrix operand order repaired and checked by reading, not compiled, not run on hardware
//
#include <hip/hip_runtime.h>
#include <hip/hip_bf16.h>

// Problem sizes (fixed by the reference)
#define V_SZ 32000
#define S_SZ 2048
#define D_SZ 1024
#define P_SZ 1024
#define B_SZ 2
#define M_TOT (B_SZ * S_SZ)   // 4096 token rows

typedef __attribute__((ext_vector_type(8)))  float  v8f;
typedef __attribute__((ext_vector_type(16))) __bf16 v16bf;
typedef __attribute__((ext_vector_type(8)))  __bf16 bf16x8;
typedef __attribute__((ext_vector_type(4)))  int    v4i;

// GEMM tiling: block 128x128, wave tile 32x64 (2x4 WMMA), BK=32
#define GBM 128
#define GBN 128
#define GBK 32
#define LDSS 40   // LDS row stride in bf16 units (80B, 16B-aligned chunks, padded)

// ---------------------------------------------------------------------------
// CDNA5 async global->LDS copy (ASYNCcnt). Builtin confirmed on this
// toolchain: (v4i AS1*, v4i AS3*, imm offset, imm cpol).
// ---------------------------------------------------------------------------
typedef __attribute__((address_space(1))) v4i as1_v4i;  // global
typedef __attribute__((address_space(3))) v4i as3_v4i;  // LDS

__device__ __forceinline__ void async_cp16(const __bf16* g, __bf16* l) {
#if __has_builtin(__builtin_amdgcn_global_load_async_to_lds_b128)
  __builtin_amdgcn_global_load_async_to_lds_b128((as1_v4i*)g, (as3_v4i*)l, 0, 0);
#else
  unsigned lofs = (unsigned)(size_t)(__attribute__((address_space(3))) __bf16*)l;
  asm volatile("global_load_async_to_lds_b128 %0, %1, off"
               :: "v"(lofs), "v"(g) : "memory");
#endif
}

__device__ __forceinline__ void wait_async0() {
#if __has_builtin(__builtin_amdgcn_s_wait_asynccnt)
  __builtin_amdgcn_s_wait_asynccnt(0);
#else
  asm volatile("s_wait_asynccnt 0" ::: "memory");
#endif
}

// ---------------------------------------------------------------------------
// reductions (wave32)
// ---------------------------------------------------------------------------
__device__ __forceinline__ float warp_max_f(float v) {
#pragma unroll
  for (int o = 16; o > 0; o >>= 1) v = fmaxf(v, __shfl_xor(v, o, 32));
  return v;
}
__device__ __forceinline__ float warp_sum_f(float v) {
#pragma unroll
  for (int o = 16; o > 0; o >>= 1) v += __shfl_xor(v, o, 32);
  return v;
}
__device__ __forceinline__ float block_max_f(float v, float* red) {
  v = warp_max_f(v);
  const int w = threadIdx.x >> 5;
  if ((threadIdx.x & 31) == 0) red[w] = v;
  __syncthreads();
  if (threadIdx.x < 8) {
    float t = red[threadIdx.x];
#pragma unroll
    for (int o = 4; o > 0; o >>= 1) t = fmaxf(t, __shfl_xor(t, o, 32));
    if (threadIdx.x == 0) red[0] = t;
  }
  __syncthreads();
  float r = red[0];
  __syncthreads();
  return r;
}
__device__ __forceinline__ float block_sum_f(float v, float* red) {
  v = warp_sum_f(v);
  const int w = threadIdx.x >> 5;
  if ((threadIdx.x & 31) == 0) red[w] = v;
  __syncthreads();
  if (threadIdx.x < 8) {
    float t = red[threadIdx.x];
#pragma unroll
    for (int o = 4; o > 0; o >>= 1) t += __shfl_xor(t, o, 32);
    if (threadIdx.x == 0) red[0] = t;
  }
  __syncthreads();
  float r = red[0];
  __syncthreads();
  return r;
}

// ---------------------------------------------------------------------------
// f32 -> bf16 elementwise convert (grid-stride)
// ---------------------------------------------------------------------------
__global__ __launch_bounds__(256) void cvt_f32_bf16(const float* __restrict__ x,
                                                    __bf16* __restrict__ y,
                                                    long long n) {
  long long i = (long long)blockIdx.x * blockDim.x + threadIdx.x;
  const long long stride = (long long)gridDim.x * blockDim.x;
  for (; i < n; i += stride) y[i] = (__bf16)x[i];
}

// ---------------------------------------------------------------------------
// f32 [R,C] -> bf16 [C,R] transpose+convert via 32x33 LDS tile; batched
// ---------------------------------------------------------------------------
__global__ __launch_bounds__(256) void transpose_cvt_kernel(
    const float* __restrict__ src, __bf16* __restrict__ dst, int R, int C,
    long long sSrc, long long sDst) {
  __shared__ float t[32][33];
  src += (long long)blockIdx.z * sSrc;
  dst += (long long)blockIdx.z * sDst;
  const int c0 = blockIdx.x * 32, r0 = blockIdx.y * 32;
  const int tx = threadIdx.x & 31, ty = threadIdx.x >> 5;  // ty 0..7
#pragma unroll
  for (int i = 0; i < 4; ++i)
    t[ty + i * 8][tx] = src[(size_t)(r0 + ty + i * 8) * C + c0 + tx];
  __syncthreads();
#pragma unroll
  for (int i = 0; i < 4; ++i)
    dst[(size_t)(c0 + ty + i * 8) * R + r0 + tx] = (__bf16)t[tx][ty + i * 8];
}

// ---------------------------------------------------------------------------
// embedding gather + positional add + LayerNorm1 -> bf16 activations
// ---------------------------------------------------------------------------
__global__ __launch_bounds__(256) void embed_ln_kernel(
    const int* __restrict__ tok, const float* __restrict__ te,
    const float* __restrict__ pe, const float* __restrict__ g,
    const float* __restrict__ b, __bf16* __restrict__ Y) {
  __shared__ float r1[8], r2[8];
  const int row = blockIdx.x;
  const int pos = row & (S_SZ - 1);
  const int t = tok[row];
  const float* xe = te + (size_t)t * D_SZ;
  const float* xp = pe + (size_t)pos * D_SZ;
  const int tid = threadIdx.x;
  float x[4], s = 0.f, ss = 0.f;
#pragma unroll
  for (int i = 0; i < 4; ++i) {
    const int d = tid + i * 256;
    const float v = xe[d] + xp[d];
    x[i] = v; s += v; ss += v * v;
  }
  s = block_sum_f(s, r1);
  ss = block_sum_f(ss, r2);
  const float mean = s * (1.0f / D_SZ);
  const float var = ss * (1.0f / D_SZ) - mean * mean;
  const float inv = rsqrtf(var + 1e-5f);
#pragma unroll
  for (int i = 0; i < 4; ++i) {
    const int d = tid + i * 256;
    Y[(size_t)row * D_SZ + d] = (__bf16)((x[i] - mean) * inv * g[d] + b[d]);
  }
}

// ---------------------------------------------------------------------------
// LayerNorm over f32 rows -> bf16 (one block per row, D=1024)
// ---------------------------------------------------------------------------
__global__ __launch_bounds__(256) void layernorm_bf16_kernel(
    const float* __restrict__ X, const float* __restrict__ g,
    const float* __restrict__ b, __bf16* __restrict__ Y) {
  __shared__ float r1[8], r2[8];
  const int row = blockIdx.x;
  const float* xr = X + (size_t)row * D_SZ;
  const int tid = threadIdx.x;
  float x[4], s = 0.f, ss = 0.f;
#pragma unroll
  for (int i = 0; i < 4; ++i) {
    const int d = tid + i * 256;
    const float v = xr[d];
    x[i] = v; s += v; ss += v * v;
  }
  s = block_sum_f(s, r1);
  ss = block_sum_f(ss, r2);
  const float mean = s * (1.0f / D_SZ);
  const float var = ss * (1.0f / D_SZ) - mean * mean;
  const float inv = rsqrtf(var + 1e-5f);
#pragma unroll
  for (int i = 0; i < 4; ++i) {
    const int d = tid + i * 256;
    Y[(size_t)row * D_SZ + d] = (__bf16)((x[i] - mean) * inv * g[d] + b[d]);
  }
}

// ---------------------------------------------------------------------------
// row softmax with scale, f32 in -> bf16 probabilities out
// ---------------------------------------------------------------------------
__global__ __launch_bounds__(256) void softmax_bf16_kernel(
    const float* __restrict__ Sc, __bf16* __restrict__ Pout, int ncol,
    float scale) {
  __shared__ float r1[8];
  const size_t row = blockIdx.x;
  const float* x = Sc + row * (size_t)ncol;
  __bf16* y = Pout + row * (size_t)ncol;
  const int tid = threadIdx.x;
  float mx = -3.0e38f;
  for (int c = tid; c < ncol; c += 256) mx = fmaxf(mx, x[c] * scale);
  mx = block_max_f(mx, r1);
  float sum = 0.f;
  for (int c = tid; c < ncol; c += 256) sum += __expf(x[c] * scale - mx);
  sum = block_sum_f(sum, r1);
  const float inv = 1.0f / sum;
  for (int c = tid; c < ncol; c += 256)
    y[c] = (__bf16)(__expf(x[c] * scale - mx) * inv);
}

// ---------------------------------------------------------------------------
// NT bf16 GEMM: C[M,N] = A[M,K] * Bsrc[N,K]^T, f32 accumulate.
// 256 threads / 8 waves. Block tile 128x128, wave tile 32x64 (2x4 WMMA),
// BK=32. Double-buffered LDS; tile k+1 streamed with async global->LDS
// copies (ASYNCcnt) while 8 WMMAs consume tile k. Optional bias; batch grid.z.
// All dims are exact multiples of the tiles for this problem.
// ---------------------------------------------------------------------------
__global__ __launch_bounds__(256) void gemm_nt_bf16_kernel(
    const __bf16* __restrict__ A, const __bf16* __restrict__ Bsrc,
    float* __restrict__ C, const float* __restrict__ bias, int M, int N, int K,
    long long strideA, long long strideB, long long strideC) {
  __shared__ __bf16 As[2][GBM * LDSS];   // (m, k)
  __shared__ __bf16 Bs[2][GBN * LDSS];   // (n, k)  == B^T tile

  const int bz = blockIdx.z;
  A += (long long)bz * strideA;
  Bsrc += (long long)bz * strideB;
  C += (long long)bz * strideC;

  const int bm = blockIdx.y * GBM;
  const int bn = blockIdx.x * GBN;
  const int tid = threadIdx.x;
  const int lane = tid & 31;
  const int wave = tid >> 5;
  const int wm = wave & 3;    // 4 row groups of 32
  const int wn = wave >> 2;   // 2 col groups of 64
  const int l16 = lane & 15;
  const int lhalf = lane >> 4;

  // staging: 16B chunks; A and B are each 128x32 -> 512 chunks (2/thread each)
  const int srow = tid >> 2;        // 0..63
  const int scol = (tid & 3) * 8;   // 0,8,16,24

  auto stage = [&](int buf, int k0) {
    async_cp16(A + (size_t)(bm + srow) * K + k0 + scol,
               &As[buf][srow * LDSS + scol]);
    async_cp16(A + (size_t)(bm + 64 + srow) * K + k0 + scol,
               &As[buf][(64 + srow) * LDSS + scol]);
    async_cp16(Bsrc + (size_t)(bn + srow) * K + k0 + scol,
               &Bs[buf][srow * LDSS + scol]);
    async_cp16(Bsrc + (size_t)(bn + 64 + srow) * K + k0 + scol,
               &Bs[buf][(64 + srow) * LDSS + scol]);
  };

  v8f acc[2][4] = {};
  const int nk = K / GBK;
  stage(0, 0);
  int cur = 0;

  for (int kt = 0; kt < nk; ++kt) {
    wait_async0();
    __syncthreads();                       // tile `cur` visible to all waves
    if (kt + 1 < nk) stage(cur ^ 1, (kt + 1) * GBK);

    const unsigned* As32 = (const unsigned*)&As[cur][0];
    const unsigned* Bs32 = (const unsigned*)&Bs[cur][0];

    union { v16bf v; unsigned u[8]; } af[2], bfr[4];
#pragma unroll
    for (int tm = 0; tm < 2; ++tm) {
      const int am = wm * 32 + tm * 16 + l16;
#pragma unroll
      for (int j = 0; j < 8; ++j) {
        // A 16x32 bf16 layout: lanes 0-15 K{0..7,16..23}; 16-31 K{8..15,24..31}
        const int kb = (j < 4 ? 2 * j : 16 + 2 * (j - 4)) + (lhalf ? 8 : 0);
        af[tm].u[j] = As32[(am * LDSS + kb) >> 1];
      }
    }
#pragma unroll
    for (int tn = 0; tn < 4; ++tn) {
      const int bnl = wn * 64 + tn * 16 + l16;
#pragma unroll
      for (int j = 0; j < 8; ++j) {
        // B 32x16 bf16 layout: lanes 0-15 K{0..7,16..23}; 16-31 K{8..15,24..31}
        const int kb = (j < 4 ? 2 * j : 16 + 2 * (j - 4)) + (lhalf ? 8 : 0);
        bfr[tn].u[j] = Bs32[(bnl * LDSS + kb) >> 1];
      }
    }
#pragma unroll
    for (int tm = 0; tm < 2; ++tm)
#pragma unroll
      for (int tn = 0; tn < 4; ++tn)
        acc[tm][tn] = __builtin_amdgcn_wmma_f32_16x16x32_bf16(
            false, af[tm].v, false, bfr[tn].v, (short)0, acc[tm][tn], false,
            false);

    cur ^= 1;
  }

  // C 16x16 f32 layout: VGPR r -> lanes 0-15 (M=r), lanes 16-31 (M=8+r)
#pragma unroll
  for (int tm = 0; tm < 2; ++tm) {
    const int crow0 = bm + wm * 32 + tm * 16 + lhalf * 8;
#pragma unroll
    for (int tn = 0; tn < 4; ++tn) {
      const int ccol = bn + wn * 64 + tn * 16 + l16;
      const float bv = bias ? bias[ccol] : 0.0f;
#pragma unroll
      for (int r = 0; r < 8; ++r)
        C[(size_t)(crow0 + r) * N + ccol] = acc[tm][tn][r] + bv;
    }
  }
}

// ---------------------------------------------------------------------------
// host
// ---------------------------------------------------------------------------
extern "C" void kernel_launch(void* const* d_in, const int* in_sizes, int n_in,
                              void* d_out, int out_size, void* d_ws,
                              size_t ws_size, hipStream_t stream) {
  (void)in_sizes; (void)n_in; (void)out_size; (void)ws_size;

  const int* tok = (const int*)d_in[0];
  const float* te = (const float*)d_in[1];
  const float* pe = (const float*)d_in[2];
  const float* pk = (const float*)d_in[3];
  const float* pv = (const float*)d_in[4];
  const float* fw = (const float*)d_in[5];
  const float* fb = (const float*)d_in[6];
  const float* ln1g = (const float*)d_in[7];
  const float* ln1b = (const float*)d_in[8];
  const float* ln2g = (const float*)d_in[9];
  const float* ln2b = (const float*)d_in[10];
  const float* ln3g = (const float*)d_in[11];
  const float* ln3b = (const float*)d_in[12];
  const float* ln4g = (const float*)d_in[13];
  const float* ln4b = (const float*)d_in[14];
  float* out = (float*)d_out;

  char* w = (char*)d_ws;
  auto carve = [&](size_t bytes) {
    char* p = w;
    w += (bytes + 255) & ~(size_t)255;
    return p;
  };
  __bf16* Kb = (__bf16*)carve((size_t)P_SZ * D_SZ * 2);         // p_key bf16 [P,D]
  __bf16* Vt = (__bf16*)carve((size_t)P_SZ * D_SZ * 2);         // p_value^T bf16 [D,P]
  __bf16* Wb = (__bf16*)carve((size_t)V_SZ * D_SZ * 2);         // fc_w bf16 [V,D]
  __bf16* Abuf = (__bf16*)carve((size_t)M_TOT * D_SZ * 2);      // normed acts [M,D]
  __bf16* Qbuf = (__bf16*)carve((size_t)M_TOT * D_SZ * 2);      // Q bf16 [B,S,D]
  __bf16* Qt = (__bf16*)carve((size_t)M_TOT * D_SZ * 2);        // Q^T bf16 [B,D,S]
  float* Sbuf = (float*)carve((size_t)B_SZ * S_SZ * S_SZ * 4);  // scores f32
  __bf16* Pbuf = (__bf16*)carve((size_t)B_SZ * S_SZ * S_SZ * 2);// probs bf16
  float* Obuf = (float*)carve((size_t)M_TOT * D_SZ * 4);        // f32 outputs

  const float scale = 0.03125f;  // 1/sqrt(1024)

  auto cvt = [&](const float* src, __bf16* dst, long long n) {
    cvt_f32_bf16<<<(int)((n + 255) / 256), 256, 0, stream>>>(src, dst, n);
  };
  auto gemm = [&](const __bf16* A, const __bf16* Bm, float* C,
                  const float* bias, int M, int N, int K, long long sA,
                  long long sB, long long sC, int nb) {
    gemm_nt_bf16_kernel<<<dim3(N / GBN, M / GBM, nb), 256, 0, stream>>>(
        A, Bm, C, bias, M, N, K, sA, sB, sC);
  };

  // one-time weight preprocessing
  cvt(pk, Kb, (long long)P_SZ * D_SZ);
  cvt(fw, Wb, (long long)V_SZ * D_SZ);
  transpose_cvt_kernel<<<dim3(D_SZ / 32, P_SZ / 32, 1), 256, 0, stream>>>(
      pv, Vt, P_SZ, D_SZ, 0, 0);

  auto pattention = [&](const __bf16* Xin, float* Oout) {
    // scores = X @ p_key^T   (M=4096, N=P, K=D)
    gemm(Xin, Kb, Sbuf, nullptr, M_TOT, P_SZ, D_SZ, 0, 0, 0, 1);
    softmax_bf16_kernel<<<M_TOT, 256, 0, stream>>>(Sbuf, Pbuf, P_SZ, scale);
    // out = probs @ p_value = probs @ (p_value^T)^T   (M=4096, N=D, K=P)
    gemm(Pbuf, Vt, Oout, nullptr, M_TOT, D_SZ, P_SZ, 0, 0, 0, 1);
  };

  // 1) embedding + pos + LN1
  embed_ln_kernel<<<M_TOT, 256, 0, stream>>>(tok, te, pe, ln1g, ln1b, Abuf);

  // 2) Q = pattention(X_norm); keep bf16 Q and Q^T
  pattention(Abuf, Obuf);
  cvt(Obuf, Qbuf, (long long)M_TOT * D_SZ);
  transpose_cvt_kernel<<<dim3(D_SZ / 32, S_SZ / 32, B_SZ), 256, 0, stream>>>(
      Obuf, Qt, S_SZ, D_SZ, (long long)S_SZ * D_SZ, (long long)S_SZ * D_SZ);

  // 3) self-attention (Q=K=V), batched over B
  gemm(Qbuf, Qbuf, Sbuf, nullptr, S_SZ, S_SZ, D_SZ, (long long)S_SZ * D_SZ,
       (long long)S_SZ * D_SZ, (long long)S_SZ * S_SZ, B_SZ);
  softmax_bf16_kernel<<<B_SZ * S_SZ, 256, 0, stream>>>(Sbuf, Pbuf, S_SZ, scale);
  // Xatt = attn @ V = attn @ (Q^T)^T   (M=S, N=D, K=S)
  gemm(Pbuf, Qt, Obuf, nullptr, S_SZ, D_SZ, S_SZ, (long long)S_SZ * S_SZ,
       (long long)S_SZ * D_SZ, (long long)S_SZ * D_SZ, B_SZ);

  // 4) LN2 -> pattention -> LN3 -> pattention -> LN4
  layernorm_bf16_kernel<<<M_TOT, 256, 0, stream>>>(Obuf, ln2g, ln2b, Abuf);
  pattention(Abuf, Obuf);
  layernorm_bf16_kernel<<<M_TOT, 256, 0, stream>>>(Obuf, ln3g, ln3b, Abuf);
  pattention(Abuf, Obuf);
  layernorm_bf16_kernel<<<M_TOT, 256, 0, stream>>>(Obuf, ln4g, ln4b, Abuf);

  // 5) logits = Xffn_norm @ fc_w^T + fc_b   (M=4096, N=32000, K=1024)
  gemm(Abuf, Wb, out, fb, M_TOT, V_SZ, D_SZ, 0, 0, 0, 1);
}
